// QuerySAT_81801947119872
// MI455X (gfx1250) — matrix operand order repaired and checked by reading, not compiled, not run on hardware
//
#include <hip/hip_runtime.h>
#include <hip/hip_bf16.h>
#include <math.h>

typedef __attribute__((ext_vector_type(16))) _Float16 v16h;
typedef __attribute__((ext_vector_type(8)))  _Float16 v8h;
typedef __attribute__((ext_vector_type(8)))  float    v8f;
typedef __attribute__((ext_vector_type(4)))  float    v4f;

// ---------------- problem constants ----------------
static constexpr long NV   = 65536;
static constexpr long NC   = 262144;
static constexpr long NG   = 128;
static constexpr long FM   = 128;   // FEATURE_MAPS
static constexpr long QM   = 128;   // QUERY_MAPS
static constexpr long LM   = 8;     // LOGIT_MAPS
static constexpr long NE   = NC * 3;
static constexpr long TWOV = 2 * NV;
static constexpr int  H_Q  = 153;   // int(128*1.2)
static constexpr int  H_C  = 204;   // int(128*1.6)
static constexpr int  H_U  = 230;   // int(128*1.8)
static constexpr int  ROUNDS = 16;

// ---------------- workspace layout (floats) ----------------
static constexpr long o_vars  = 0;
static constexpr long o_cs    = o_vars  + NV * FM;
static constexpr long o_v1    = o_cs    + NC * FM;
static constexpr long o_qh    = o_v1    + NV * (FM + 4);
static constexpr long o_query = o_qh    + NV * H_Q;
static constexpr long o_lits  = o_query + NV * QM;
static constexpr long o_cl    = o_lits  + TWOV * QM;
static constexpr long o_gsum  = o_cl    + NC * QM;
static constexpr long o_vgrad = o_gsum  + TWOV * QM;
static constexpr long o_cunit = o_vgrad + NV * QM;
static constexpr long o_ch    = o_cunit + NC * (FM + QM);
static constexpr long o_cdata = o_ch    + NC * H_C;
static constexpr long o_cd2   = o_cdata + NC * (FM + QM);
static constexpr long o_vloss = o_cd2   + NC * FM;
static constexpr long o_unit  = o_vloss + TWOV * QM;
static constexpr long o_uh1   = o_unit  + NV * (FM + 3 * QM);
static constexpr long o_uh2   = o_uh1   + NV * H_U;
static constexpr long o_nvars = o_uh2   + NV * H_U;
static constexpr long o_logit = o_nvars + NV * FM;
static constexpr long o_lits8 = o_logit + NV * LM;
static constexpr long o_cl8   = o_lits8 + TWOV * LM;
static constexpr long o_pg    = o_cl8   + NC * LM;
static constexpr long o_sums  = o_pg    + NG * LM;
static constexpr long o_gvar  = o_sums  + NG * FM;
static constexpr long o_cntv  = o_gvar  + NG;
static constexpr long o_cntc  = o_cntv  + NG;
static constexpr long o_ldeg  = o_cntc  + NG;
static constexpr long o_degw  = o_ldeg  + TWOV;
static constexpr long o_vdegw = o_degw  + TWOV;
static constexpr long o_loss  = o_vdegw + NV;
static constexpr long o_best  = o_loss  + 1;   // 128 ints

// ---------------- small helpers ----------------
__device__ __forceinline__ float softplusf(float x) {
  if (x > 15.f)  return x;
  if (x < -15.f) return __expf(x);
  return log1pf(__expf(x));
}

#define GRID_STRIDE(i, n) \
  for (long i = (long)blockIdx.x * blockDim.x + threadIdx.x; i < (n); \
       i += (long)gridDim.x * blockDim.x)

// ================= WMMA tiled GEMM =================
// C[M,N] = act(A[M,K] @ W[K,N] + bias[N]);  f32 in/out, f16 WMMA, f32 accum.
// Block: 256 threads = 8 wave32.  Block tile 128(M) x 128(N), K step 32,
// double-buffered LDS.  M must be a multiple of 128 (true for all callers).
// Wave (wm,wn) in 4x2 grid owns a 32x64 patch -> 8 accumulators of 16x16.
__global__ __launch_bounds__(256) void k_gemm(
    const float* __restrict__ A, const float* __restrict__ W,
    const float* __restrict__ bias, float* __restrict__ C,
    int K, int N, int relu) {
  __shared__ __align__(16) _Float16 sA[2][128][40];  // [row][k]
  __shared__ __align__(16) _Float16 sB[2][128][40];  // [col][k] (transposed!)

  const int tid  = threadIdx.x;
  const int wave = tid >> 5;
  const int lane = tid & 31;
  const int wm   = wave >> 1;           // 0..3  (M sub-tile, 32 rows)
  const int wn   = wave & 1;            // 0..1  (N sub-tile, 64 cols)
  const int half = lane >> 4;
  const int l15  = lane & 15;
  const long m0  = (long)blockIdx.x * 128;
  const long n0  = (long)blockIdx.y * 128;

  const int ar = tid >> 1, ac0 = (tid & 1) * 16;     // A: row, 16 consecutive k
  const int bc = tid & 127, bk0 = (tid >> 7) * 16;   // B: col, 16 consecutive k
  const bool fullN = (n0 + 128 <= N);

  v8f acc[2][4];
  #pragma unroll
  for (int mt = 0; mt < 2; ++mt)
    #pragma unroll
    for (int nt = 0; nt < 4; ++nt) acc[mt][nt] = {};

  // stage K-slab [k0, k0+32) into buffer `buf`
  auto stage = [&](int buf, int k0) {
    // ---- A tile: 128 rows x 32 k ----
    _Float16 av[16];
    if (k0 + 32 <= K) {
      const float* Ap = A + (m0 + ar) * (long)K + k0 + ac0;
      #pragma unroll
      for (int j = 0; j < 16; ++j) av[j] = (_Float16)Ap[j];
      if (k0 + 64 < K) __builtin_prefetch(Ap + 32, 0, 1);
    } else {
      // tail: clamp address (always valid load), select value -> no exec games
      #pragma unroll
      for (int j = 0; j < 16; ++j) {
        int kk = k0 + ac0 + j;
        int kc = kk < K ? kk : K - 1;
        float x = A[(m0 + ar) * (long)K + kc];
        av[j] = (_Float16)(kk < K ? x : 0.f);
      }
    }
    v8h a0, a1;
    #pragma unroll
    for (int j = 0; j < 8; ++j) { a0[j] = av[j]; a1[j] = av[8 + j]; }
    *(v8h*)&sA[buf][ar][ac0]     = a0;
    *(v8h*)&sA[buf][ar][ac0 + 8] = a1;

    // ---- B tile: 32 k x 128 n, stored transposed [col][k] ----
    _Float16 bv[16];
    if ((k0 + 32 <= K) && fullN) {
      const float* Wp = W + (long)(k0 + bk0) * N + n0 + bc;
      #pragma unroll
      for (int j = 0; j < 16; ++j) bv[j] = (_Float16)Wp[(long)j * N];
      if (k0 + 64 < K) __builtin_prefetch(Wp + 32 * (long)N, 0, 1);
    } else {
      long cc = (n0 + bc < N) ? (n0 + bc) : (N - 1);
      #pragma unroll
      for (int j = 0; j < 16; ++j) {
        int kk = k0 + bk0 + j;
        int kc = kk < K ? kk : K - 1;
        float x = W[(long)kc * N + cc];
        bv[j] = (_Float16)((kk < K && n0 + bc < N) ? x : 0.f);
      }
    }
    v8h b0, b1;
    #pragma unroll
    for (int j = 0; j < 8; ++j) { b0[j] = bv[j]; b1[j] = bv[8 + j]; }
    *(v8h*)&sB[buf][bc][bk0]     = b0;
    *(v8h*)&sB[buf][bc][bk0 + 8] = b1;
  };

  stage(0, 0);
  __syncthreads();
  int buf = 0;
  for (int k0 = 0; k0 < K; k0 += 32) {
    if (k0 + 32 < K) stage(buf ^ 1, k0 + 32);  // overlap next slab with WMMA

    v16h af[2], bf[4];
    #pragma unroll
    for (int t = 0; t < 2; ++t) {
      const _Float16* pa = &sA[buf][wm * 32 + t * 16 + l15][half * 8];
      v8h alo = *(const v8h*)pa;
      v8h ahi = *(const v8h*)(pa + 16);
      af[t] = __builtin_shufflevector(alo, ahi, 0, 1, 2, 3, 4, 5, 6, 7, 8, 9,
                                      10, 11, 12, 13, 14, 15);
    }
    #pragma unroll
    for (int t = 0; t < 4; ++t) {
      const _Float16* pb = &sB[buf][wn * 64 + t * 16 + l15][half * 8];
      v8h blo = *(const v8h*)pb;
      v8h bhi = *(const v8h*)(pb + 16);
      bf[t] = __builtin_shufflevector(blo, bhi, 0, 1, 2, 3, 4, 5, 6, 7, 8, 9,
                                      10, 11, 12, 13, 14, 15);
    }
    #pragma unroll
    for (int mt = 0; mt < 2; ++mt)
      #pragma unroll
      for (int nt = 0; nt < 4; ++nt)
        acc[mt][nt] = __builtin_amdgcn_wmma_f32_16x16x32_f16(
            false, af[mt], false, bf[nt], (short)0, acc[mt][nt], false, false);
    __syncthreads();
    buf ^= 1;
  }

  // store: C-layout VGPR i -> row (i + 8*half), col = lane&15
  #pragma unroll
  for (int nt = 0; nt < 4; ++nt) {
    long col = n0 + wn * 64 + nt * 16 + l15;
    float b = (fullN || col < N) ? bias[col] : 0.f;
    #pragma unroll
    for (int mt = 0; mt < 2; ++mt) {
      float* Cp = C + (m0 + wm * 32 + mt * 16 + half * 8) * (long)N + col;
      if (fullN || col < N) {
        #pragma unroll
        for (int i = 0; i < 8; ++i) {
          float v = acc[mt][nt][i] + b;
          if (relu) v = fmaxf(v, 0.f);
          Cp[(long)i * N] = v;
        }
      }
    }
  }
}

// ================= elementwise / sparse kernels =================
__global__ void k_fill(float* p, float v, long n) { GRID_STRIDE(i, n) p[i] = v; }

// vectorized column-block copy; all callers have lds%4==0, ldd%4==0, cols%4==0
__global__ void k_copy4(float* dst, long ldd, const float* src, long lds,
                        long rows, long cols, float scale) {
  long c4 = cols >> 2;
  GRID_STRIDE(i, rows * c4) {
    long r = i / c4, c = (i - r * c4) << 2;
    v4f v = *(const v4f*)&src[r * lds + c];
    *(v4f*)&dst[r * ldd + c] = v * scale;
  }
}

__global__ void k_softplus_pm(const float* q, float* lits, long n) {
  GRID_STRIDE(i, n) {
    float x = q[i];
    lits[i]     = softplusf(x);
    lits[n + i] = softplusf(-x);
  }
}

// out[c,f] = exp(-(lits[i0,f]+lits[i1,f]+lits[i2,f])), float4 over features
__global__ void k_clause_exp4(const float* lits, const int* lit_idx, float* out,
                              int F, long total4) {
  int F4 = F >> 2;
  GRID_STRIDE(i, total4) {
    long c = i / F4; int f = (int)(i - c * F4) << 2;
    long e0 = 3 * c;
    v4f s = *(const v4f*)&lits[(long)lit_idx[e0] * F + f] +
            *(const v4f*)&lits[(long)lit_idx[e0 + 1] * F + f] +
            *(const v4f*)&lits[(long)lit_idx[e0 + 2] * F + f];
    v4f o;
    #pragma unroll
    for (int j = 0; j < 4; ++j) o[j] = __expf(-s[j]);
    *(v4f*)&out[c * F + f] = o;
  }
}

// acc[lit_idx[e], f..f+3] += src[e/3, f..f+3]
__global__ void k_edge_scatter4(const float* src, int ldc, const int* lit_idx,
                                float* acc, int F, long total4) {
  int F4 = F >> 2;
  GRID_STRIDE(i, total4) {
    long e = i / F4; int f = (int)(i - e * F4) << 2;
    long c = e / 3;
    v4f v = *(const v4f*)&src[c * (long)ldc + f];
    long base = (long)lit_idx[e] * F + f;
    atomicAdd(&acc[base + 0], v.x);
    atomicAdd(&acc[base + 1], v.y);
    atomicAdd(&acc[base + 2], v.z);
    atomicAdd(&acc[base + 3], v.w);
  }
}

__global__ void k_vgrad(const float* q, const float* gsum, const float* vdegw,
                        float* vg, long n) {
  GRID_STRIDE(i, n) {
    long v = i >> 7;
    float x  = q[i];
    float sp = 1.f / (1.f + __expf(-x));
    float sn = 1.f / (1.f + __expf(x));
    vg[i] = vdegw[v] * (-gsum[i] * sp + gsum[n + i] * sn);
  }
}

__global__ void k_scale_rows(float* x, const float* w, long n) {
  GRID_STRIDE(i, n) x[i] *= w[i >> 7];
}

// per-graph feature sums via LDS-resident partials (launch with few blocks)
__global__ __launch_bounds__(256) void k_pn_sum(const float* x, const int* seg,
                                                float* sums, long rows) {
  __shared__ float loc[NG * 128];
  for (int j = threadIdx.x; j < NG * 128; j += 256) loc[j] = 0.f;
  __syncthreads();
  GRID_STRIDE(i, rows * 128) {
    long r = i >> 7;
    atomicAdd(&loc[((long)seg[r] << 7) + (i & 127)], x[i]);
  }
  __syncthreads();
  for (int j = threadIdx.x; j < NG * 128; j += 256) {
    float v = loc[j];
    if (v != 0.f) atomicAdd(&sums[j], v);
  }
}

__global__ __launch_bounds__(128) void k_pn_center(
    float* x, const int* seg, const float* sums, const float* cnt,
    float* varacc) {
  long r = blockIdx.x; int f = threadIdx.x;
  int g = seg[r];
  float c  = fmaxf(cnt[g], 1.f);
  float xc = x[r * 128 + f] - sums[g * 128 + f] / c;
  x[r * 128 + f] = xc;
  __shared__ float red[128];
  red[f] = xc * xc;
  __syncthreads();
  for (int s = 64; s > 0; s >>= 1) {
    if (f < s) red[f] += red[f + s];
    __syncthreads();
  }
  if (f == 0) atomicAdd(&varacc[g], red[0] * (1.f / 128.f));
}

__global__ void k_pn_norm(float* x, const int* seg, const float* varacc,
                          const float* cnt, long rows) {
  GRID_STRIDE(i, rows * 128) {
    long r = i >> 7; int g = seg[r];
    float c = fmaxf(cnt[g], 1.f);
    x[i] *= rsqrtf(varacc[g] / c + 1e-6f);
  }
}

__global__ void k_axpby(float* y, const float* x, float a, float b, long n) {
  GRID_STRIDE(i, n) y[i] = a * x[i] + b * y[i];
}

__global__ __launch_bounds__(256) void k_count(const int* seg, float* cnt,
                                               long n) {
  __shared__ float loc[NG];
  for (int j = threadIdx.x; j < NG; j += 256) loc[j] = 0.f;
  __syncthreads();
  GRID_STRIDE(i, n) atomicAdd(&loc[seg[i]], 1.f);
  __syncthreads();
  for (int j = threadIdx.x; j < NG; j += 256) {
    float v = loc[j];
    if (v != 0.f) atomicAdd(&cnt[j], v);
  }
}

__global__ void k_edge_deg(const int* lit_idx, float* litdeg, long n) {
  GRID_STRIDE(i, n) atomicAdd(&litdeg[lit_idx[i]], 1.f);
}

__global__ void k_degw(const float* litdeg, float* degw, float* vdegw) {
  GRID_STRIDE(i, TWOV) {
    degw[i] = rsqrtf(fmaxf(litdeg[i], 1.f));
    if (i < NV)
      vdegw[i] = 4.f * rsqrtf(fmaxf(litdeg[i] + litdeg[i + NV], 1.f));
  }
}

__global__ void k_mixed(float* v, long n) {
  GRID_STRIDE(i, n) {
    float x = v[i];
    v[i] = x * -logf(1.f - x + 1e-10f);
  }
}

// per-graph logit-map sums with LDS partials (launch with few blocks)
__global__ __launch_bounds__(256) void k_pg(const float* mixed, const int* cgid,
                                            float* pg) {
  __shared__ float loc[NG * LM];
  for (int j = threadIdx.x; j < NG * LM; j += 256) loc[j] = 0.f;
  __syncthreads();
  GRID_STRIDE(i, NC * LM) {
    long c = i >> 3;
    atomicAdd(&loc[((long)cgid[c] << 3) + (i & 7)], mixed[i]);
  }
  __syncthreads();
  for (int j = threadIdx.x; j < NG * LM; j += 256) {
    float v = loc[j];
    if (v != 0.f) atomicAdd(&pg[j], v);
  }
}

__global__ __launch_bounds__(128) void k_loss_best(const float* pg,
                                                   float* loss_acc, int* best) {
  int g = threadIdx.x;
  float v[8]; float mn = 1e30f; int mi = 0;
  #pragma unroll
  for (int j = 0; j < 8; ++j) {
    float p = pg[g * 8 + j];
    if (p < mn) { mn = p; mi = j; }
    v[j] = sqrtf(p + 1e-6f) - 1e-3f;
  }
  best[g] = mi;
  #pragma unroll
  for (int a = 1; a < 8; ++a) {  // insertion sort, descending
    float key = v[a]; int b = a - 1;
    while (b >= 0 && v[b] < key) { v[b + 1] = v[b]; --b; }
    v[b + 1] = key;
  }
  const float costs[8] = {1, 4, 9, 16, 25, 36, 49, 64};
  float acc = 0.f;
  #pragma unroll
  for (int j = 0; j < 8; ++j) acc += v[j] * costs[j];
  __shared__ float red[128];
  red[g] = acc;
  __syncthreads();
  for (int s = 64; s > 0; s >>= 1) {
    if (g < s) red[g] += red[g + s];
    __syncthreads();
  }
  if (g == 0) atomicAdd(loss_acc, red[0] * (1.f / 204.f));
}

__global__ void k_final(const float* logits, const int* best, const int* vgid,
                        const float* loss_acc, float* out) {
  GRID_STRIDE(i, NV + 1) {
    if (i < NV) out[i] = logits[i * 8 + best[vgid[i]]];
    else        out[NV] = loss_acc[0] * (1.f / (float)ROUNDS);
  }
}

// ================= orchestration =================
static inline dim3 gs(long n) {
  long b = (n + 255) / 256;
  if (b > 1048576) b = 1048576;
  return dim3((unsigned)b);
}

extern "C" void kernel_launch(void* const* d_in, const int* in_sizes, int n_in,
                              void* d_out, int out_size, void* d_ws,
                              size_t ws_size, hipStream_t stream) {
  // params flattened in setup_inputs() dict order: query, clause, update, output
  const float* qw0 = (const float*)d_in[0];  const float* qb0 = (const float*)d_in[1];
  const float* qw1 = (const float*)d_in[2];  const float* qb1 = (const float*)d_in[3];
  const float* cw0 = (const float*)d_in[4];  const float* cb0 = (const float*)d_in[5];
  const float* cw1 = (const float*)d_in[6];  const float* cb1 = (const float*)d_in[7];
  const float* uw0 = (const float*)d_in[8];  const float* ub0 = (const float*)d_in[9];
  const float* uw1 = (const float*)d_in[10]; const float* ub1 = (const float*)d_in[11];
  const float* uw2 = (const float*)d_in[12]; const float* ub2 = (const float*)d_in[13];
  const float* ow0 = (const float*)d_in[14]; const float* ob0 = (const float*)d_in[15];
  const float* ow1 = (const float*)d_in[16]; const float* ob1 = (const float*)d_in[17];
  const float* noise    = (const float*)d_in[18];
  const int*   lit_idx  = (const int*)d_in[19];
  const int*   vgid     = (const int*)d_in[21];
  const int*   cgid     = (const int*)d_in[22];

  float* ws  = (float*)d_ws;
  float* out = (float*)d_out;

  auto gemm = [&](const float* A, const float* W, const float* B, float* C,
                  long M, int K, int N, int relu) {
    dim3 grid((unsigned)(M / 128), (unsigned)((N + 127) / 128));
    k_gemm<<<grid, 256, 0, stream>>>(A, W, B, C, K, N, relu);
  };
  auto zero = [&](long off, long n) {
    k_fill<<<gs(n), 256, 0, stream>>>(ws + off, 0.f, n);
  };
  auto copy4 = [&](long dof, long ldd, const float* src, long lds, long rows,
                   long cols, float scale) {
    k_copy4<<<gs(rows * (cols >> 2)), 256, 0, stream>>>(ws + dof, ldd, src, lds,
                                                        rows, cols, scale);
  };

  // ---- one-time init ----
  k_fill<<<gs(NV * FM), 256, 0, stream>>>(ws + o_vars, 1.f, NV * FM);
  k_fill<<<gs(NC * FM), 256, 0, stream>>>(ws + o_cs, 1.f, NC * FM);
  zero(o_ldeg, TWOV); zero(o_cntv, NG); zero(o_cntc, NG); zero(o_loss, 1);
  k_edge_deg<<<gs(NE), 256, 0, stream>>>(lit_idx, ws + o_ldeg, NE);
  k_count<<<dim3(128), 256, 0, stream>>>(vgid, ws + o_cntv, NV);
  k_count<<<dim3(128), 256, 0, stream>>>(cgid, ws + o_cntc, NC);
  k_degw<<<gs(TWOV), 256, 0, stream>>>(ws + o_ldeg, ws + o_degw, ws + o_vdegw);

  for (int t = 0; t < ROUNDS; ++t) {
    const float* noise_t = noise + (long)t * NV * 4;

    // v1 = concat(vars, noise_t)  -> query MLP
    copy4(o_v1, FM + 4, ws + o_vars, FM, NV, FM, 1.f);
    copy4(o_v1 + FM, FM + 4, noise_t, 4, NV, 4, 1.f);
    gemm(ws + o_v1, qw0, qb0, ws + o_qh, NV, FM + 4, H_Q, 1);
    gemm(ws + o_qh, qw1, qb1, ws + o_query, NV, H_Q, (int)QM, 0);

    // softplus loss + its gradient wrt query
    k_softplus_pm<<<gs(NV * QM), 256, 0, stream>>>(ws + o_query, ws + o_lits, NV * QM);
    k_clause_exp4<<<gs(NC * (QM >> 2)), 256, 0, stream>>>(ws + o_lits, lit_idx, ws + o_cl, (int)QM, NC * (QM >> 2));
    zero(o_gsum, TWOV * QM);
    k_edge_scatter4<<<gs(NE * (QM >> 2)), 256, 0, stream>>>(ws + o_cl, (int)QM, lit_idx, ws + o_gsum, (int)QM, NE * (QM >> 2));
    k_vgrad<<<gs(NV * QM), 256, 0, stream>>>(ws + o_query, ws + o_gsum, ws + o_vdegw, ws + o_vgrad, NV * QM);

    // clause MLP on concat(clause_state, 4*clauses_loss)
    copy4(o_cunit, FM + QM, ws + o_cs, FM, NC, FM, 1.f);
    copy4(o_cunit + FM, FM + QM, ws + o_cl, QM, NC, QM, 4.f);
    gemm(ws + o_cunit, cw0, cb0, ws + o_ch, NC, (int)(FM + QM), H_C, 1);
    gemm(ws + o_ch, cw1, cb1, ws + o_cdata, NC, H_C, (int)(FM + QM), 0);

    // pair_norm(clause_data[:, QM:]) ; clause_state = pn + 0.1*clause_state
    copy4(o_cd2, FM, ws + o_cdata + QM, FM + QM, NC, FM, 1.f);
    zero(o_sums, NG * FM); zero(o_gvar, NG);
    k_pn_sum<<<dim3(512), 256, 0, stream>>>(ws + o_cd2, cgid, ws + o_sums, NC);
    k_pn_center<<<dim3((unsigned)NC), 128, 0, stream>>>(ws + o_cd2, cgid, ws + o_sums, ws + o_cntc, ws + o_gvar);
    k_pn_norm<<<gs(NC * FM), 256, 0, stream>>>(ws + o_cd2, cgid, ws + o_gvar, ws + o_cntc, NC);
    k_axpby<<<gs(NC * FM), 256, 0, stream>>>(ws + o_cs, ws + o_cd2, 1.f, 0.1f, NC * FM);

    // variables_loss = adj_mm(clause_data[:, :QM]) * degree_weight
    zero(o_vloss, TWOV * QM);
    k_edge_scatter4<<<gs(NE * (QM >> 2)), 256, 0, stream>>>(ws + o_cdata, (int)(FM + QM), lit_idx, ws + o_vloss, (int)QM, NE * (QM >> 2));
    k_scale_rows<<<gs(TWOV * QM), 256, 0, stream>>>(ws + o_vloss, ws + o_degw, TWOV * QM);

    // unit = concat(vgrad, vars, vloss[:NV], vloss[NV:]) -> update MLP
    copy4(o_unit, 512, ws + o_vgrad, QM, NV, QM, 1.f);
    copy4(o_unit + 128, 512, ws + o_vars, FM, NV, FM, 1.f);
    copy4(o_unit + 256, 512, ws + o_vloss, QM, NV, QM, 1.f);
    copy4(o_unit + 384, 512, ws + o_vloss + NV * QM, QM, NV, QM, 1.f);
    gemm(ws + o_unit, uw0, ub0, ws + o_uh1, NV, 512, H_U, 1);
    gemm(ws + o_uh1, uw1, ub1, ws + o_uh2, NV, H_U, H_U, 1);
    gemm(ws + o_uh2, uw2, ub2, ws + o_nvars, NV, H_U, (int)FM, 0);

    // pair_norm over variables ; vars = pn + 0.1*vars
    zero(o_sums, NG * FM); zero(o_gvar, NG);
    k_pn_sum<<<dim3(512), 256, 0, stream>>>(ws + o_nvars, vgid, ws + o_sums, NV);
    k_pn_center<<<dim3((unsigned)NV), 128, 0, stream>>>(ws + o_nvars, vgid, ws + o_sums, ws + o_cntv, ws + o_gvar);
    k_pn_norm<<<gs(NV * FM), 256, 0, stream>>>(ws + o_nvars, vgid, ws + o_gvar, ws + o_cntv, NV);
    k_axpby<<<gs(NV * FM), 256, 0, stream>>>(ws + o_vars, ws + o_nvars, 1.f, 0.1f, NV * FM);

    // output MLP -> logits ; per-graph sorted loss + argmin
    gemm(ws + o_vars, ow0, ob0, ws + o_uh1, NV, (int)FM, (int)FM, 1);
    gemm(ws + o_uh1, ow1, ob1, ws + o_logit, NV, (int)FM, (int)LM, 0);
    k_softplus_pm<<<gs(NV * LM), 256, 0, stream>>>(ws + o_logit, ws + o_lits8, NV * LM);
    k_clause_exp4<<<gs(NC * (LM >> 2)), 256, 0, stream>>>(ws + o_lits8, lit_idx, ws + o_cl8, (int)LM, NC * (LM >> 2));
    k_mixed<<<gs(NC * LM), 256, 0, stream>>>(ws + o_cl8, NC * LM);
    zero(o_pg, NG * LM);
    k_pg<<<dim3(256), 256, 0, stream>>>(ws + o_cl8, cgid, ws + o_pg);
    k_loss_best<<<1, 128, 0, stream>>>(ws + o_pg, ws + o_loss, (int*)(ws + o_best));
  }

  k_final<<<gs(NV + 1), 256, 0, stream>>>(ws + o_logit, (int*)(ws + o_best),
                                          vgid, ws + o_loss, out);
}
